// AttentionLayerO3TwoUpdateNodeGeneral_2259152797797
// MI455X (gfx1250) — matrix operand order repaired and checked by reading, not compiled, not run on hardware
//
#include <hip/hip_runtime.h>
#include <hip/hip_bf16.h>
#include <hip/hip_fp16.h>

typedef __attribute__((ext_vector_type(16))) _Float16 v16h;
typedef __attribute__((ext_vector_type(8)))  float    v8f;

#define DEVINL __device__ __forceinline__

// ---------------- problem constants ----------------
constexpr int Nn   = 8192;
constexpr int Ee   = 131072;
constexpr int Tt   = 160000;
constexpr int KVIN = 456;       // 3*(4+20) + 3*128
constexpr int KVPAD= 480;       // padded K for GEMM1 (15 k-tiles of 32)
constexpr int KVS  = 488;       // LDS row stride in halfs (244 dw, 244%64=52 -> conflict free)

// ---------------- WMMA helpers ----------------
DEVINL v8f wmma16x16x32(v16h a, v16h b, v8f c) {
  // D = A(16x32 f16) * B(32x16 f16) + C(16x16 f32)
  return __builtin_amdgcn_wmma_f32_16x16x32_f16(false, a, false, b, (short)0, c, false, false);
}

// A fragment (16x32 f16) from a row-major tile (stride in halfs), per ISA layout:
// lanes 0-15: m=lane, K = {0..7, 16..23}; lanes 16-31: m=lane-16, K = {8..15, 24..31}
DEVINL v16h load_A(const _Float16* X, int stride, int k0, int lane) {
  const int m  = lane & 15;
  const int kb = (lane & 16) ? 8 : 0;
  const _Float16* p = X + m * stride + k0 + kb;
  v16h a;
#pragma unroll
  for (int vg = 0; vg < 4; ++vg) {
    a[2*vg+0]   = p[2*vg+0];
    a[2*vg+1]   = p[2*vg+1];
    a[8+2*vg+0] = p[16+2*vg+0];
    a[8+2*vg+1] = p[16+2*vg+1];
  }
  return a;
}

// CDNA5 async global->LDS copy of one 16B unit (per active lane), ASYNCcnt-tracked.
DEVINL void async_copy_b128(void* ldsDst, const void* gsrc) {
  // low 32 bits of a flat LDS address are the wave-relative LDS offset
  unsigned ldsOff = (unsigned)(uintptr_t)ldsDst;
  asm volatile("global_load_async_to_lds_b128 %0, %1, off"
               :: "v"(ldsOff), "v"(gsrc) : "memory");
}
DEVINL void wait_async_all() {
  asm volatile("s_wait_asynccnt 0" ::: "memory");
}

// One wave computes a 16xN tile: C = A(16 x KT*32) * B + bias, result to LDS (f32).
// B is pre-swizzled (k_pack_b): fragment (kt,nt) = 32 lanes x 16 contiguous halfs.
template<int KT>
DEVINL void gemm_tile(const _Float16* aSrc, int aStride,
                      const _Float16* __restrict__ Bp, int NT,
                      const float* __restrict__ bias,
                      float* outLds, int outStride, int lane) {
  v16h aReg[KT];
#pragma unroll
  for (int kt = 0; kt < KT; ++kt) aReg[kt] = load_A(aSrc, aStride, kt * 32, lane);
  const int col16 = lane & 15;
  const int mb    = (lane & 16) ? 8 : 0;
  for (int nt = 0; nt < NT; ++nt) {
    v8f acc = {0.f,0.f,0.f,0.f,0.f,0.f,0.f,0.f};
#pragma unroll
    for (int kt = 0; kt < KT; ++kt) {
      v16h b = *(const v16h*)(Bp + ((size_t)(kt * NT + nt) * 32 + lane) * 16);
      acc = wmma16x16x32(aReg[kt], b, acc);
    }
    const int   col = nt * 16 + col16;
    const float bv  = bias[col];
#pragma unroll
    for (int vg = 0; vg < 8; ++vg)
      outLds[(mb + vg) * outStride + col] = acc[vg] + bv;
  }
}

// Wave-cooperative LayerNorm(128) + ReLU over a 16x128 fp32 tile (stride 132),
// result as f16 tile (stride 136). stats: 32 floats scratch.
DEVINL void ln_relu(const float* hid, const float* __restrict__ g, const float* __restrict__ bt,
                    _Float16* nrm, float* stats, int lane) {
  __syncthreads();
  if (lane < 16) {
    const float* row = hid + lane * 132;
    float m = 0.f;
    for (int c = 0; c < 128; ++c) m += row[c];
    m *= (1.f / 128.f);
    float v = 0.f;
    for (int c = 0; c < 128; ++c) { float d = row[c] - m; v += d * d; }
    v *= (1.f / 128.f);
    stats[lane]      = m;
    stats[16 + lane] = rsqrtf(v + 1e-5f);
  }
  __syncthreads();
  for (int idx = lane; idx < 2048; idx += 32) {
    int r = idx >> 7, c = idx & 127;
    float val = (hid[r * 132 + c] - stats[r]) * stats[16 + r] * g[c] + bt[c];
    nrm[r * 136 + c] = (_Float16)fmaxf(val, 0.f);
  }
  __syncthreads();
}

DEVINL void atomicMaxF(float* addr, float val) {
  if (val >= 0.f) atomicMax((int*)addr, __float_as_int(val));
  else            atomicMin((unsigned int*)addr, (unsigned int)__float_as_int(val));
}

DEVINL float sigmoidf(float x) { return 1.f / (1.f + expf(-x)); }

// ---------------- small prep kernels ----------------

// Per-edge: rel_x, gaussian smearing, f16 edge chunk [ef(4) | r_feat(20)], edge weights.
__global__ void k_edge(const float* __restrict__ x, const float* __restrict__ ef,
                       const int* __restrict__ eidx,
                       const float* __restrict__ ewW_h2x, const float* __restrict__ ewb_h2x,
                       const float* __restrict__ ewW_x2h, const float* __restrict__ ewb_x2h,
                       _Float16* __restrict__ edgepack, float* __restrict__ relx,
                       float* __restrict__ ewX, float* __restrict__ ewH) {
  int e = blockIdx.x * 256 + threadIdx.x;
  if (e >= Ee) return;
  int s = eidx[e], d = eidx[Ee + e];
  float dx = x[d*3+0] - x[s*3+0];
  float dy = x[d*3+1] - x[s*3+1];
  float dz = x[d*3+2] - x[s*3+2];
  float dist = sqrtf(dx*dx + dy*dy + dz*dz);
  relx[e*3+0] = dx; relx[e*3+1] = dy; relx[e*3+2] = dz;
  const float step  = 10.f / 19.f;
  const float coeff = -0.5f / (step * step);
  _Float16* pk = edgepack + (size_t)e * 24;
  pk[0] = (_Float16)ef[e*4+0]; pk[1] = (_Float16)ef[e*4+1];
  pk[2] = (_Float16)ef[e*4+2]; pk[3] = (_Float16)ef[e*4+3];
  float accX = ewb_x2h[0], accH = ewb_h2x[0];
#pragma unroll
  for (int g = 0; g < 20; ++g) {
    float t  = dist - (float)g * step;
    float rf = expf(coeff * t * t);
    pk[4 + g] = (_Float16)rf;
    accX += rf * ewW_x2h[g];
    accH += rf * ewW_h2x[g];
  }
  ewX[e] = sigmoidf(accX);
  ewH[e] = sigmoidf(accH);
}

__global__ void k_f2h(const float* __restrict__ src, _Float16* __restrict__ dst, int n) {
  int i = blockIdx.x * 256 + threadIdx.x;
  if (i < n) dst[i] = (_Float16)src[i];
}

// Pack fp32 (Ksrc x N, row major) weight into swizzled f16 B fragments with K zero-padded to Kpad.
__global__ void k_pack_b(const float* __restrict__ src, _Float16* __restrict__ dst,
                         int Ksrc, int Kpad, int Ncol) {
  int idx = blockIdx.x * 256 + threadIdx.x;
  int total = (Kpad / 32) * (Ncol / 16) * 512;
  if (idx >= total) return;
  int j    = idx & 15;
  int lane = (idx >> 4) & 31;
  int tile = idx >> 9;
  int NT   = Ncol >> 4;
  int nt   = tile % NT, kt = tile / NT;
  int k = kt * 32 + ((lane & 16) ? 8 : 0) + (j & 7) + ((j & 8) ? 16 : 0);
  int n = nt * 16 + (lane & 15);
  float v = (k < Ksrc) ? src[(size_t)k * Ncol + n] : 0.f;
  dst[idx] = (_Float16)v;
}

__global__ void k_init(float* m1x, float* m2x, float* s1x, float* s2x,
                       float* m1h, float* m2h, float* s1h, float* s2h,
                       float* outAcc, float* delta) {
  int idx = blockIdx.x * 256 + threadIdx.x;
  if (idx < Nn * 16) {
    m1x[idx] = -1e30f; m2x[idx] = -1e30f; s1x[idx] = 0.f; s2x[idx] = 0.f;
    m1h[idx] = -1e30f; m2h[idx] = -1e30f; s1h[idx] = 0.f; s2h[idx] = 0.f;
  }
  if (idx < Nn * 128) outAcc[idx] = 0.f;
  if (idx < Nn * 3)   delta[idx]  = 0.f;
}

// ---------------- node-level fused MLP (q MLPs, out MLP) ----------------
// one wave per 16-row tile; input is [in0 | in1] (each N x 128 fp32), Kin = KT*32.
template<int KT>
__global__ void __launch_bounds__(32)
k_node_mlp(const float* __restrict__ in0, const float* __restrict__ in1,
           const _Float16* __restrict__ W1p, const float* __restrict__ b1,
           const float* __restrict__ g1, const float* __restrict__ bt1,
           const _Float16* __restrict__ W2p, const float* __restrict__ b2,
           const float* __restrict__ res, float* __restrict__ outF32,
           _Float16* __restrict__ outF16) {
  constexpr int Kin  = KT * 32;
  constexpr int ASTR = Kin + 8;
  extern __shared__ char smem[];
  _Float16* aLds  = (_Float16*)smem;                 // 16 x ASTR
  float*    hid   = (float*)(aLds + 16 * ASTR);      // 16 x 132
  float*    stats = hid + 16 * 132;                  // 32
  _Float16* nrm   = (_Float16*)(stats + 32);         // 16 x 136
  const int lane = threadIdx.x;
  const int row0 = blockIdx.x * 16;

  for (int idx = lane; idx < 16 * Kin; idx += 32) {
    int r = idx / Kin, c = idx - r * Kin;
    float v;
    if (KT == 4 || c < 128) v = in0[(size_t)(row0 + r) * 128 + (c & 127)];
    else                    v = in1[(size_t)(row0 + r) * 128 + (c - 128)];
    aLds[r * ASTR + c] = (_Float16)v;
  }
  __syncthreads();

  gemm_tile<KT>(aLds, ASTR, W1p, 8, b1, hid, 132, lane);
  ln_relu(hid, g1, bt1, nrm, stats, lane);

  v16h aReg[4];
#pragma unroll
  for (int kt = 0; kt < 4; ++kt) aReg[kt] = load_A(nrm, 136, kt * 32, lane);
  const int col16 = lane & 15;
  const int mb    = (lane & 16) ? 8 : 0;
  for (int nt = 0; nt < 8; ++nt) {
    v8f acc = {0.f,0.f,0.f,0.f,0.f,0.f,0.f,0.f};
#pragma unroll
    for (int kt = 0; kt < 4; ++kt) {
      v16h b = *(const v16h*)(W2p + ((size_t)(kt * 8 + nt) * 32 + lane) * 16);
      acc = wmma16x16x32(aReg[kt], b, acc);
    }
    const int c = nt * 16 + col16;
#pragma unroll
    for (int vg = 0; vg < 8; ++vg) {
      int row = row0 + mb + vg;
      float val = acc[vg] + b2[c];
      if (res)    val += res[(size_t)row * 128 + c];
      outF32[(size_t)row * 128 + c] = val;
      if (outF16) outF16[(size_t)row * 128 + c] = (_Float16)val;
    }
  }
}

// ---------------- triplet MLP + attention epilogue ----------------
// MODE 0 = x2h (v out 256 -> vcomb T x 128), MODE 1 = h2x (v out 32 -> cji/cki T x 16)
template<int MODE>
__global__ void __launch_bounds__(128)
k_triplet(const _Float16* __restrict__ hsrc, const _Float16* __restrict__ edgepack,
          const int* __restrict__ id_i, const int* __restrict__ id_j, const int* __restrict__ id_k,
          const int* __restrict__ e_ji, const int* __restrict__ e_ki, const int* __restrict__ e_kj,
          const _Float16* __restrict__ Wk1, const float* __restrict__ kb1,
          const float* __restrict__ kg1,  const float* __restrict__ kbt1,
          const _Float16* __restrict__ Wk2, const float* __restrict__ kb2,
          const _Float16* __restrict__ Wv1, const float* __restrict__ vb1,
          const float* __restrict__ vg1,  const float* __restrict__ vbt1,
          const _Float16* __restrict__ Wv2, const float* __restrict__ vb2,
          const float* __restrict__ qsrc, const float* __restrict__ ew,
          float* __restrict__ e1g, float* __restrict__ e2g,
          float* __restrict__ m1g, float* __restrict__ m2g,
          float* __restrict__ aux1, float* __restrict__ aux2) {
  constexpr int NTV = (MODE == 0) ? 16 : 2;
  constexpr int VW  = NTV * 16;
  extern __shared__ char smem[];
  _Float16* kv    = (_Float16*)smem;                  // 2 x 16 x KVS
  int*      idsL  = (int*)(kv + 2 * 16 * KVS);        // 6 x 16
  float*    stats = (float*)(idsL + 96);              // 4 x 32
  _Float16* nrm   = (_Float16*)(stats + 4 * 32);      // 4 x 16 x 136
  // union region: fp32 hidden (GEMM1 phase) overlaps GEMM2 outputs (epilogue phase)
  float*    hid   = (float*)(nrm + 4 * 16 * 136);     // 4 x 16 x 132
  float*    kOut  = hid;                              // 2 x 16 x 128
  float*    vOut  = kOut + 2 * 16 * 128;              // 2 x 16 x VW

  const int tid  = threadIdx.x;
  const int lane = tid & 31;
  const int wav  = tid >> 5;
  const int t0   = blockIdx.x * 16;

  __builtin_prefetch(Wk1, 0, 3);   // global_prefetch: warm L2->WGP path for weights
  __builtin_prefetch(Wv1, 0, 3);

  if (tid < 96) {
    const int* srcs[6] = {id_i, id_j, id_k, e_ji, e_ki, e_kj};
    idsL[tid] = srcs[tid >> 4][t0 + (tid & 15)];
  }
  __syncthreads();

  // Build both permuted KV tiles with CDNA5 async global->LDS DMA.
  // 57 16B-units per row: units 0..8 -> edge chunks (24 halfs each),
  // units 9..56 -> three 128-half h rows.
  for (int u = tid; u < 2 * 16 * 57; u += 128) {
    int var  = u / (16 * 57);
    int rem  = u - var * (16 * 57);
    int r    = rem / 57;
    int unit = rem - r * 57;
    const void* src;
    if (unit < 9) {
      int c = unit / 3, o = unit - c * 3;
      int which = (var == 0) ? c : ((c == 0) ? 1 : ((c == 1) ? 0 : 2)); // kv2 swaps ji<->ki
      int eid = idsL[(3 + which) * 16 + r];
      src = (const void*)(((const uint4*)(edgepack + (size_t)eid * 24)) + o);
    } else {
      int uu = unit - 9;
      int c = uu >> 4, o = uu & 15;
      int which = (var == 0) ? c : ((c == 0) ? 0 : ((c == 1) ? 2 : 1)); // kv2 swaps hj<->hk
      int nid = idsL[which * 16 + r];
      src = (const void*)(((const uint4*)(hsrc + (size_t)nid * 128)) + o);
    }
    void* dst = (void*)(kv + (size_t)var * 16 * KVS + (size_t)r * KVS + (size_t)unit * 8);
    async_copy_b128(dst, src);
  }
  wait_async_all();
  __syncthreads();

  // wave -> (kv variant, mlp: 0=k, 1=v); uniform code path so barriers stay legal
  const int var = wav & 1;
  const int mlp = wav >> 1;
  const _Float16* A1     = kv + (size_t)var * 16 * KVS;
  float*          hidW   = hid + wav * 16 * 132;
  float*          statsW = stats + wav * 32;
  _Float16*       nrmW   = nrm + wav * 16 * 136;

  const _Float16* W1  = mlp ? Wv1  : Wk1;
  const float*    b1  = mlp ? vb1  : kb1;
  const float*    g1  = mlp ? vg1  : kg1;
  const float*    bt1 = mlp ? vbt1 : kbt1;
  const _Float16* W2  = mlp ? Wv2  : Wk2;
  const float*    b2  = mlp ? vb2  : kb2;
  const int       NT2 = mlp ? NTV  : 8;

  gemm_tile<15>(A1, KVS, W1, 8, b1, hidW, 132, lane);          // 16 x 480 x 128
  ln_relu(hidW, g1, bt1, nrmW, statsW, lane);                  // ends with barrier -> hid dead
  float*    outW      = mlp ? (vOut + (size_t)var * 16 * VW) : (kOut + (size_t)var * 16 * 128);
  const int outStride = mlp ? VW : 128;
  gemm_tile<4>(nrmW, 136, W2, NT2, b2, outW, outStride, lane); // 16 x 128 x Nout
  __syncthreads();

  // epilogue: attention logits + value combination
  const float scale = 0.3535533906f; // 1/sqrt(8)
  for (int it = tid; it < 256; it += 128) {
    int r = it >> 4, hh = it & 15;
    int tt = t0 + r;
    int i  = idsL[r];
    const float* q   = qsrc + (size_t)i * 128 + hh * 8;
    const float* kk1 = kOut + (size_t)r * 128 + hh * 8;
    const float* kk2 = kOut + 16 * 128 + (size_t)r * 128 + hh * 8;
    float e1 = 0.f, e2 = 0.f;
#pragma unroll
    for (int d = 0; d < 8; ++d) { e1 += kk1[d] * q[d]; e2 += kk2[d] * q[d]; }
    e1 *= scale; e2 *= scale;
    e1g[(size_t)tt * 16 + hh] = e1;
    e2g[(size_t)tt * 16 + hh] = e2;
    atomicMaxF(m1g + (size_t)i * 16 + hh, e1);
    atomicMaxF(m2g + (size_t)i * 16 + hh, e2);
  }
  if (MODE == 0) {
    for (int it = tid; it < 16 * 128; it += 128) {
      int r = it >> 7, c = it & 127;
      int tt = t0 + r;
      float ewji = ew[idsL[3 * 16 + r]];
      float ewki = ew[idsL[4 * 16 + r]];
      float v = 0.5f * ((vOut[(size_t)r * 256 + c]       + vOut[16 * 256 + (size_t)r * 256 + c])       * ewji
                      + (vOut[(size_t)r * 256 + 128 + c] + vOut[16 * 256 + (size_t)r * 256 + 128 + c]) * ewki);
      aux1[(size_t)tt * 128 + c] = v;
    }
  } else {
    for (int it = tid; it < 256; it += 128) {
      int r = it >> 4, hh = it & 15;
      int tt = t0 + r;
      float ewji = ew[idsL[3 * 16 + r]];
      float ewki = ew[idsL[4 * 16 + r]];
      float cji = 0.5f * (vOut[(size_t)r * 32 + hh]      + vOut[16 * 32 + (size_t)r * 32 + hh])      * ewji;
      float cki = 0.5f * (vOut[(size_t)r * 32 + 16 + hh] + vOut[16 * 32 + (size_t)r * 32 + 16 + hh]) * ewki;
      aux1[(size_t)tt * 16 + hh] = cji;
      aux2[(size_t)tt * 16 + hh] = cki;
    }
  }
}

// ---------------- softmax / scatter kernels ----------------
__global__ void k_expsum(const float* __restrict__ e1, const float* __restrict__ e2,
                         const float* __restrict__ m1, const float* __restrict__ m2,
                         float* __restrict__ s1, float* __restrict__ s2,
                         const int* __restrict__ ids) {
  int idx = blockIdx.x * 256 + threadIdx.x;
  if (idx >= Tt * 16) return;
  int tt = idx >> 4, hh = idx & 15;
  size_t ih = (size_t)ids[tt] * 16 + hh;
  atomicAdd(&s1[ih], expf(e1[idx] - m1[ih]));
  atomicAdd(&s2[ih], expf(e2[idx] - m2[ih]));
}

__global__ void k_scatter_x2h(const float* __restrict__ e1, const float* __restrict__ e2,
                              const float* __restrict__ m1, const float* __restrict__ m2,
                              const float* __restrict__ s1, const float* __restrict__ s2,
                              const float* __restrict__ vcomb, const int* __restrict__ ids,
                              float* __restrict__ outAcc) {
  int idx = blockIdx.x * 256 + threadIdx.x;
  if (idx >= Tt * 16) return;
  int tt = idx >> 4, hh = idx & 15;
  int i = ids[tt];
  size_t ih = (size_t)i * 16 + hh;
  float a = 0.5f * (expf(e1[idx] - m1[ih]) / s1[ih] + expf(e2[idx] - m2[ih]) / s2[ih]);
  const float* v = vcomb + (size_t)tt * 128 + hh * 8;
  float* o = outAcc + (size_t)i * 128 + hh * 8;
#pragma unroll
  for (int d = 0; d < 8; ++d) atomicAdd(o + d, a * v[d]);
}

__global__ void k_scatter_h2x(const float* __restrict__ e1, const float* __restrict__ e2,
                              const float* __restrict__ m1, const float* __restrict__ m2,
                              const float* __restrict__ s1, const float* __restrict__ s2,
                              const float* __restrict__ cji, const float* __restrict__ cki,
                              const float* __restrict__ relx, const int* __restrict__ ids,
                              const int* __restrict__ e_ji, const int* __restrict__ e_ki,
                              float* __restrict__ delta) {
  int idx = blockIdx.x * 256 + threadIdx.x;
  if (idx >= Tt * 16) return;
  int tt = idx >> 4, hh = idx & 15;
  int i = ids[tt];
  size_t ih = (size_t)i * 16 + hh;
  float a = 0.5f * (expf(e1[idx] - m1[ih]) / s1[ih] + expf(e2[idx] - m2[ih]) / s2[ih]);
  float cj = a * cji[idx] * (1.f / 16.f);
  float ck = a * cki[idx] * (1.f / 16.f);
  const float* rj = relx + (size_t)e_ji[tt] * 3;
  const float* rk = relx + (size_t)e_ki[tt] * 3;
#pragma unroll
  for (int c = 0; c < 3; ++c)
    atomicAdd(&delta[(size_t)i * 3 + c], cj * rj[c] + ck * rk[c]);
}

__global__ void k_axpy(const float* __restrict__ x, const float* __restrict__ d,
                       float* __restrict__ out, int n) {
  int i = blockIdx.x * 256 + threadIdx.x;
  if (i < n) out[i] = x[i] + d[i];
}

// ---------------- host launch ----------------
extern "C" void kernel_launch(void* const* d_in, const int* in_sizes, int n_in,
                              void* d_out, int out_size, void* d_ws, size_t ws_size,
                              hipStream_t stream) {
  (void)in_sizes; (void)n_in; (void)out_size; (void)ws_size;
  // ---- inputs (JAX pytree flatten: sorted dict keys; 'h2x' < 'x2h') ----
  const float* h   = (const float*)d_in[0];
  const float* x   = (const float*)d_in[1];
  const float* ef  = (const float*)d_in[2];
  const int* eidx  = (const int*)d_in[3];
  const int* id_i  = (const int*)d_in[4];
  const int* id_j  = (const int*)d_in[5];
  const int* id_k  = (const int*)d_in[6];
  const int* e_ji  = (const int*)d_in[7];
  const int* e_ki  = (const int*)d_in[8];
  const int* e_kj  = (const int*)d_in[9];
  // h2x: ew_W, ew_b, k{W1,W2,b1,b2,bt1,g1}, q{...}, v{...}
  const float* h2x_ewW = (const float*)d_in[10];
  const float* h2x_ewb = (const float*)d_in[11];
  const float *hk_W1=(const float*)d_in[12], *hk_W2=(const float*)d_in[13],
              *hk_b1=(const float*)d_in[14], *hk_b2=(const float*)d_in[15],
              *hk_bt1=(const float*)d_in[16], *hk_g1=(const float*)d_in[17];
  const float *hq_W1=(const float*)d_in[18], *hq_W2=(const float*)d_in[19],
              *hq_b1=(const float*)d_in[20], *hq_b2=(const float*)d_in[21],
              *hq_bt1=(const float*)d_in[22], *hq_g1=(const float*)d_in[23];
  const float *hv_W1=(const float*)d_in[24], *hv_W2=(const float*)d_in[25],
              *hv_b1=(const float*)d_in[26], *hv_b2=(const float*)d_in[27],
              *hv_bt1=(const float*)d_in[28], *hv_g1=(const float*)d_in[29];
  // x2h: ew_W, ew_b, k, out, q, v
  const float* x2h_ewW = (const float*)d_in[30];
  const float* x2h_ewb = (const float*)d_in[31];
  const float *xk_W1=(const float*)d_in[32], *xk_W2=(const float*)d_in[33],
              *xk_b1=(const float*)d_in[34], *xk_b2=(const float*)d_in[35],
              *xk_bt1=(const float*)d_in[36], *xk_g1=(const float*)d_in[37];
  const float *xo_W1=(const float*)d_in[38], *xo_W2=(const float*)d_in[39],
              *xo_b1=(const float*)d_in[40], *xo_b2=(const float*)d_in[41],
              *xo_bt1=(const float*)d_in[42], *xo_g1=(const float*)d_in[43];
  const float *xq_W1=(const float*)d_in[44], *xq_W2=(const float*)d_in[45],
              *xq_b1=(const float*)d_in[46], *xq_b2=(const float*)d_in[47],
              *xq_bt1=(const float*)d_in[48], *xq_g1=(const float*)d_in[49];
  const float *xv_W1=(const float*)d_in[50], *xv_W2=(const float*)d_in[51],
              *xv_b1=(const float*)d_in[52], *xv_b2=(const float*)d_in[53],
              *xv_bt1=(const float*)d_in[54], *xv_g1=(const float*)d_in[55];

  // ---- workspace carve-up ----
  size_t cur = 0;
  auto alloc = [&](size_t bytes) -> void* {
    cur = (cur + 255) & ~(size_t)255;
    void* p = (char*)d_ws + cur;
    cur += bytes;
    return p;
  };
  _Float16* hF16    = (_Float16*)alloc((size_t)Nn * 128 * 2);
  _Float16* houtF16 = (_Float16*)alloc((size_t)Nn * 128 * 2);
  _Float16* edgepack= (_Float16*)alloc((size_t)Ee * 24 * 2);
  float* relx   = (float*)alloc((size_t)Ee * 3 * 4);
  float* ewX    = (float*)alloc((size_t)Ee * 4);
  float* ewH    = (float*)alloc((size_t)Ee * 4);
  float* qX     = (float*)alloc((size_t)Nn * 128 * 4);
  float* qH     = (float*)alloc((size_t)Nn * 128 * 4);
  float* outAcc = (float*)alloc((size_t)Nn * 128 * 4);
  // packed f16 weights
  _Float16* WkX1 = (_Float16*)alloc((size_t)KVPAD*128*2);
  _Float16* WvX1 = (_Float16*)alloc((size_t)KVPAD*128*2);
  _Float16* WkH1 = (_Float16*)alloc((size_t)KVPAD*128*2);
  _Float16* WvH1 = (_Float16*)alloc((size_t)KVPAD*128*2);
  _Float16* WkX2 = (_Float16*)alloc((size_t)128*128*2);
  _Float16* WvX2 = (_Float16*)alloc((size_t)128*256*2);
  _Float16* WkH2 = (_Float16*)alloc((size_t)128*128*2);
  _Float16* WvH2 = (_Float16*)alloc((size_t)128*32*2);
  _Float16* Wq1X = (_Float16*)alloc((size_t)128*128*2);
  _Float16* Wq2X = (_Float16*)alloc((size_t)128*128*2);
  _Float16* Wq1H = (_Float16*)alloc((size_t)128*128*2);
  _Float16* Wq2H = (_Float16*)alloc((size_t)128*128*2);
  _Float16* Wo1  = (_Float16*)alloc((size_t)256*128*2);
  _Float16* Wo2  = (_Float16*)alloc((size_t)128*128*2);
  // attention intermediates
  float* e1x = (float*)alloc((size_t)Tt*16*4);
  float* e2x = (float*)alloc((size_t)Tt*16*4);
  float* e1h = (float*)alloc((size_t)Tt*16*4);
  float* e2h = (float*)alloc((size_t)Tt*16*4);
  float* m1x = (float*)alloc((size_t)Nn*16*4);
  float* m2x = (float*)alloc((size_t)Nn*16*4);
  float* s1x = (float*)alloc((size_t)Nn*16*4);
  float* s2x = (float*)alloc((size_t)Nn*16*4);
  float* m1h = (float*)alloc((size_t)Nn*16*4);
  float* m2h = (float*)alloc((size_t)Nn*16*4);
  float* s1h = (float*)alloc((size_t)Nn*16*4);
  float* s2h = (float*)alloc((size_t)Nn*16*4);
  float* vcomb = (float*)alloc((size_t)Tt*128*4);
  float* cji   = (float*)alloc((size_t)Tt*16*4);
  float* cki   = (float*)alloc((size_t)Tt*16*4);
  float* delta = (float*)alloc((size_t)Nn*3*4);

  float* outH  = (float*)d_out;                  // h_out (N x 128)
  float* outXp = (float*)d_out + (size_t)Nn*128; // x + delta (N x 3)

  // ---- dynamic LDS sizes (hid unions with kOut+vOut) ----
  const size_t tripFixed = 2*16*KVS*2 + 96*4 + 4*32*4 + 4*16*136*2;
  const size_t hidBytes  = (size_t)4*16*132*4;
  const size_t smemTrip0 = tripFixed + ((hidBytes > (size_t)(2*16*128 + 2*16*256)*4)
                                        ? hidBytes : (size_t)(2*16*128 + 2*16*256)*4);
  const size_t smemTrip1 = tripFixed + ((hidBytes > (size_t)(2*16*128 + 2*16*32)*4)
                                        ? hidBytes : (size_t)(2*16*128 + 2*16*32)*4);
  const size_t smemNode4 = 16*(128+8)*2 + 16*132*4 + 32*4 + 16*136*2;
  const size_t smemNode8 = 16*(256+8)*2 + 16*132*4 + 32*4 + 16*136*2;
  hipFuncSetAttribute(reinterpret_cast<const void*>(&k_triplet<0>),
                      hipFuncAttributeMaxDynamicSharedMemorySize, (int)smemTrip0);
  hipFuncSetAttribute(reinterpret_cast<const void*>(&k_triplet<1>),
                      hipFuncAttributeMaxDynamicSharedMemorySize, (int)smemTrip1);

  // ---- stage 0: prep ----
  k_edge<<<(Ee + 255) / 256, 256, 0, stream>>>(x, ef, eidx, h2x_ewW, h2x_ewb,
                                               x2h_ewW, x2h_ewb, edgepack, relx, ewX, ewH);
  k_f2h<<<(Nn * 128 + 255) / 256, 256, 0, stream>>>(h, hF16, Nn * 128);

  auto pack = [&](const float* src, _Float16* dst, int Ksrc, int Kpad, int Ncol) {
    int total = (Kpad / 32) * (Ncol / 16) * 512;
    k_pack_b<<<(total + 255) / 256, 256, 0, stream>>>(src, dst, Ksrc, Kpad, Ncol);
  };
  pack(xk_W1, WkX1, KVIN, KVPAD, 128);  pack(xv_W1, WvX1, KVIN, KVPAD, 128);
  pack(hk_W1, WkH1, KVIN, KVPAD, 128);  pack(hv_W1, WvH1, KVIN, KVPAD, 128);
  pack(xk_W2, WkX2, 128, 128, 128);     pack(xv_W2, WvX2, 128, 128, 256);
  pack(hk_W2, WkH2, 128, 128, 128);     pack(hv_W2, WvH2, 128, 128, 32);
  pack(xq_W1, Wq1X, 128, 128, 128);     pack(xq_W2, Wq2X, 128, 128, 128);
  pack(hq_W1, Wq1H, 128, 128, 128);     pack(hq_W2, Wq2H, 128, 128, 128);
  pack(xo_W1, Wo1, 256, 256, 128);      pack(xo_W2, Wo2, 128, 128, 128);

  k_init<<<(Nn * 128 + 255) / 256, 256, 0, stream>>>(m1x, m2x, s1x, s2x,
                                                     m1h, m2h, s1h, s2h, outAcc, delta);

  // ---- stage 1: x2h ----
  k_node_mlp<4><<<Nn / 16, 32, smemNode4, stream>>>(h, nullptr, Wq1X, xq_b1, xq_g1, xq_bt1,
                                                    Wq2X, xq_b2, nullptr, qX, nullptr);
  k_triplet<0><<<Tt / 16, 128, smemTrip0, stream>>>(hF16, edgepack,
      id_i, id_j, id_k, e_ji, e_ki, e_kj,
      WkX1, xk_b1, xk_g1, xk_bt1, WkX2, xk_b2,
      WvX1, xv_b1, xv_g1, xv_bt1, WvX2, xv_b2,
      qX, ewX, e1x, e2x, m1x, m2x, vcomb, nullptr);
  k_expsum<<<Tt * 16 / 256, 256, 0, stream>>>(e1x, e2x, m1x, m2x, s1x, s2x, id_i);
  k_scatter_x2h<<<Tt * 16 / 256, 256, 0, stream>>>(e1x, e2x, m1x, m2x, s1x, s2x,
                                                   vcomb, id_i, outAcc);
  // out MLP on concat([attn_out, h]) + residual h -> h_out (written straight to d_out)
  k_node_mlp<8><<<Nn / 16, 32, smemNode8, stream>>>(outAcc, h, Wo1, xo_b1, xo_g1, xo_bt1,
                                                    Wo2, xo_b2, h, outH, houtF16);

  // ---- stage 2: h2x (consumes h_out) ----
  k_node_mlp<4><<<Nn / 16, 32, smemNode4, stream>>>(outH, nullptr, Wq1H, hq_b1, hq_g1, hq_bt1,
                                                    Wq2H, hq_b2, nullptr, qH, nullptr);
  k_triplet<1><<<Tt / 16, 128, smemTrip1, stream>>>(houtF16, edgepack,
      id_i, id_j, id_k, e_ji, e_ki, e_kj,
      WkH1, hk_b1, hk_g1, hk_bt1, WkH2, hk_b2,
      WvH1, hv_b1, hv_g1, hv_bt1, WvH2, hv_b2,
      qH, ewH, e1h, e2h, m1h, m2h, cji, cki);
  k_expsum<<<Tt * 16 / 256, 256, 0, stream>>>(e1h, e2h, m1h, m2h, s1h, s2h, id_i);
  k_scatter_h2x<<<Tt * 16 / 256, 256, 0, stream>>>(e1h, e2h, m1h, m2h, s1h, s2h,
                                                   cji, cki, relx, id_i, e_ji, e_ki, delta);
  k_axpy<<<(Nn * 3 + 255) / 256, 256, 0, stream>>>(x, delta, outXp, Nn * 3);
}
